// ParallelVocoderNetwork_53781580480700
// MI455X (gfx1250) — matrix operand order repaired and checked by reading, not compiled, not run on hardware
//
#include <hip/hip_runtime.h>
#include <hip/hip_bf16.h>
#include <math.h>

#define NFR   10
#define MGCS  60
#define UPS   200
#define FSZ   64
#define NB    2000
#define RFLD  1024
#define NLAY  10
#define SCL   0.70710678118f

typedef _Float16     v16h  __attribute__((ext_vector_type(16)));
typedef float        v8f   __attribute__((ext_vector_type(8)));
typedef unsigned int u32x4 __attribute__((ext_vector_type(4)));

static __device__ __forceinline__ float sigm(float x) {
    return 1.0f / (1.0f + __expf(-x));
}

// Load one 16x32 f16 A-fragment. Row data (K contiguous) -> per-lane two 16B chunks:
// dwords [s*16 + hi*4 .. +3] give K = 8*hi + 0..7, dwords [s*16+8+hi*4 .. +3] give
// K = 16 + 8*hi + 0..7, matching the ISA 16-bit A layout (CDNA5 ISA 7.12.2).
static __device__ __forceinline__ v16h load_a_frag(const _Float16* rowbase, int s, int hi) {
    union { u32x4 u[2]; v16h h; } t;
    const u32x4* p = (const u32x4*)((const unsigned int*)rowbase + s * 16 + hi * 4);
    t.u[0] = p[0];
    t.u[1] = p[2];   // +8 dwords
    return t.h;
}

// ---------------- conditioning: cond[b][m] = tanh(mgc @ cond_w.T + cond_b) ----------------
__global__ __launch_bounds__(256) void k_cond(const float* __restrict__ mgc,
                                              const float* __restrict__ cw,
                                              const float* __restrict__ cb,
                                              float* __restrict__ cond) {
    int idx = blockIdx.x * 256 + threadIdx.x;          // b*60 + m
    if (idx >= NB * MGCS) return;
    int b = idx / MGCS, m = idx % MGCS;
    int f = b / UPS,    u = b % UPS;
    int o = u * MGCS + m;
    const float* mg = mgc + f * MGCS;
    const float* w  = cw + (size_t)o * MGCS;
    float s = cb[o];
    #pragma unroll 4
    for (int k = 0; k < MGCS; ++k) s += mg[k] * w[k];
    cond[idx] = tanhf(s);
}

// -------- per-layer conditioning projections ic/gc[l][b][c] = cond @ lw.T + lb --------
__global__ __launch_bounds__(256) void k_condproj(const float* __restrict__ cond,
                                                  const float* __restrict__ lwi,
                                                  const float* __restrict__ lbi,
                                                  const float* __restrict__ lwg,
                                                  const float* __restrict__ lbg,
                                                  float* __restrict__ ic,
                                                  float* __restrict__ gc) {
    int idx = blockIdx.x * 256 + threadIdx.x;          // l*NB*64 + b*64 + c
    if (idx >= NLAY * NB * FSZ) return;
    int c  = idx & 63;
    int bl = idx >> 6;
    int b  = bl % NB;
    int l  = bl / NB;
    const float* cd = cond + b * MGCS;
    const float* wi = lwi + (size_t)(l * FSZ + c) * MGCS;
    const float* wg = lwg + (size_t)(l * FSZ + c) * MGCS;
    float si = lbi[l * FSZ + c], sg = lbg[l * FSZ + c];
    #pragma unroll 4
    for (int k = 0; k < MGCS; ++k) { si += cd[k] * wi[k]; sg += cd[k] * wg[k]; }
    ic[idx] = si;
    gc[idx] = sg;
}

// ------ one-time: swizzle all conv weights into WMMA B-fragment order, f16 ------
// blob half index (per layer): conv*8192 + nt*2048 + ks*512 + lane*16 + j
__global__ __launch_bounds__(256) void k_prepw(const float* __restrict__ cw_in,
                                               const float* __restrict__ cw_gate,
                                               const float* __restrict__ cw_res,
                                               _Float16* __restrict__ wblob) {
    int e = blockIdx.x * 256 + threadIdx.x;            // 9 layers * 3 convs * 8192
    if (e >= 9 * 3 * 8192) return;
    int l    = e / 24576;
    int r    = e % 24576;
    int conv = r >> 13;
    int q    = r & 8191;
    int j    = q & 15;
    int lane = (q >> 4) & 31;
    int ks   = (q >> 9) & 3;
    int nt   = (q >> 11) & 3;
    int k    = ks * 32 + ((j >> 3) << 4) + ((lane >> 4) << 3) + (j & 7); // global K 0..127
    int n    = nt * 16 + (lane & 15);                  // out channel
    int cin  = k & 63, kk = k >> 6;
    const float* W = (conv == 0 ? cw_in : conv == 1 ? cw_gate : cw_res)
                     + (size_t)l * FSZ * FSZ * 2;
    wblob[e] = (_Float16)W[n * 128 + cin * 2 + kk];
}

// ------ one-time: swizzle head weights pre_w[256][64] into fragment order ------
__global__ __launch_bounds__(256) void k_preph(const float* __restrict__ prew,
                                               _Float16* __restrict__ hblob) {
    int e = blockIdx.x * 256 + threadIdx.x;            // 16 nt * 2 ks * 32 * 16
    if (e >= 16384) return;
    int j    = e & 15;
    int lane = (e >> 4) & 31;
    int ks   = (e >> 9) & 1;
    int nt   = (e >> 10) & 15;
    int k    = ks * 32 + ((j >> 3) << 4) + ((lane >> 4) << 3) + (j & 7);
    int n    = nt * 16 + (lane & 15);
    hblob[nt * 1024 + ks * 512 + lane * 16 + j] = (_Float16)prew[n * 64 + k];
}

// ---------------- layer 0 (C_in = 1): scalar conv + gate, emit f16 [B][512][64] ----------------
__global__ __launch_bounds__(256) void k_layer0(const float* __restrict__ noise,
                                                const float* __restrict__ w0i,
                                                const float* __restrict__ w0g,
                                                const float* __restrict__ w0r,
                                                const float* __restrict__ bi,
                                                const float* __restrict__ bg,
                                                const float* __restrict__ br,
                                                const float* __restrict__ ic,
                                                const float* __restrict__ gc,
                                                _Float16* __restrict__ out) {
    int idx = blockIdx.x * 256 + threadIdx.x;          // b*512 + t
    if (idx >= NB * 512) return;
    int b = idx >> 9, t = idx & 511;
    float x0 = noise[b + 2 * t];
    float x1 = noise[b + 2 * t + 1];
    const float* icb = ic + b * FSZ;
    const float* gcb = gc + b * FSZ;
    _Float16 hbuf[FSZ];
    #pragma unroll
    for (int c = 0; c < FSZ; ++c) {
        float vi = w0i[2 * c] * x0 + w0i[2 * c + 1] * x1 + bi[c] + icb[c];
        float vg = w0g[2 * c] * x0 + w0g[2 * c + 1] * x1 + bg[c] + gcb[c];
        float vr = w0r[2 * c] * x0 + w0r[2 * c + 1] * x1 + br[c];
        hbuf[c] = (_Float16)((tanhf(vi) * sigm(vg) + vr) * SCL);
    }
    u32x4* dst = (u32x4*)(out + (size_t)idx * FSZ);
    const u32x4* src = (const u32x4*)hbuf;
    #pragma unroll
    for (int i = 0; i < 8; ++i) dst[i] = src[i];
}

// ---------------- layers 1..9: three fused GEMMs (in/gate/res) via WMMA ----------------
// Two 16-row M-tiles per wave: each LDS B-fragment feeds two WMMAs.
__global__ __launch_bounds__(256) void k_layer(const _Float16* __restrict__ in,
                                               _Float16* __restrict__ out,
                                               const _Float16* __restrict__ wblob,
                                               const float* __restrict__ bi,
                                               const float* __restrict__ bg,
                                               const float* __restrict__ br,
                                               const float* __restrict__ ic,
                                               const float* __restrict__ gc,
                                               int lg,     // log2(Lout)
                                               int mtiles) {
    __shared__ __align__(32) u32x4 bsh[3072];              // 48 KB of f16 fragments
    {
        const u32x4* src = (const u32x4*)wblob;
        #pragma unroll
        for (int i = 0; i < 12; ++i)
            bsh[threadIdx.x + i * 256] = src[threadIdx.x + i * 256];
    }
    __syncthreads();
    const _Float16* bf = (const _Float16*)bsh;

    int wave  = threadIdx.x >> 5;
    int lane  = threadIdx.x & 31;
    int pairs = (mtiles + 1) >> 1;
    int wid   = blockIdx.x * 8 + wave;
    if (wid >= pairs) return;                              // wave-uniform: EXEC stays full
    int hi = lane >> 4, ln = lane & 15;

    int mt0 = wid * 2;
    bool store1 = (mt0 + 1) < mtiles;
    int mt1 = store1 ? mt0 + 1 : mt0;                      // clamp: stay in-bounds

    const _Float16* arow0 = in + (size_t)(mt0 * 16 + ln) * 128;
    const _Float16* arow1 = in + (size_t)(mt1 * 16 + ln) * 128;
    __builtin_prefetch(arow0, 0, 1);                       // global_prefetch_b8
    __builtin_prefetch(arow1, 0, 1);

    v8f acc[2][3][4] = {};
    #pragma unroll
    for (int s = 0; s < 4; ++s) {
        v16h a0 = load_a_frag(arow0, s, hi);
        v16h a1 = load_a_frag(arow1, s, hi);
        #pragma unroll
        for (int c = 0; c < 3; ++c) {
            #pragma unroll
            for (int nt = 0; nt < 4; ++nt) {
                v16h b = *(const v16h*)(bf + ((c * 4 + nt) * 4 + s) * 512 + lane * 16);
                acc[0][c][nt] = __builtin_amdgcn_wmma_f32_16x16x32_f16(
                    false, a0, false, b, (short)0, acc[0][c][nt], false, false);
                acc[1][c][nt] = __builtin_amdgcn_wmma_f32_16x16x32_f16(
                    false, a1, false, b, (short)0, acc[1][c][nt], false, false);
            }
        }
    }

    // epilogue: gated activation + residual, f16 store in next-layer A layout
    int mts[2] = {mt0, mt1};
    #pragma unroll
    for (int m = 0; m < 2; ++m) {
        if (m == 1 && !store1) break;                      // wave-uniform
        #pragma unroll
        for (int nt = 0; nt < 4; ++nt) {
            int col = nt * 16 + ln;
            float cbi = bi[col], cbg = bg[col], cbr = br[col];
            #pragma unroll
            for (int r = 0; r < 8; ++r) {
                int row = mts[m] * 16 + hi * 8 + r;        // D layout: VGPR r -> M = r + 8*hi
                int b   = row >> lg;
                float vi = acc[m][0][nt][r] + cbi + ic[b * FSZ + col];
                float vg = acc[m][1][nt][r] + cbg + gc[b * FSZ + col];
                float vr = acc[m][2][nt][r] + cbr;
                out[(size_t)row * FSZ + col] =
                    (_Float16)((tanhf(vi) * sigm(vg) + vr) * SCL);
            }
        }
    }
}

// ---------------- head: pre = relu(h @ pre_w.T + pre_b); fused mean/logvar/sample ----------------
__global__ __launch_bounds__(256) void k_head(const _Float16* __restrict__ h,
                                              const _Float16* __restrict__ hblob,
                                              const float* __restrict__ preb,
                                              const float* __restrict__ meanw,
                                              const float* __restrict__ meanb,
                                              const float* __restrict__ stdw,
                                              const float* __restrict__ stdb,
                                              const float* __restrict__ noise,
                                              float* __restrict__ outp) {
    __shared__ __align__(32) u32x4 hsh[2048];              // 32 KB of f16 fragments
    {
        const u32x4* src = (const u32x4*)hblob;
        #pragma unroll
        for (int i = 0; i < 8; ++i)
            hsh[threadIdx.x + i * 256] = src[threadIdx.x + i * 256];
    }
    __syncthreads();
    const _Float16* bf = (const _Float16*)hsh;

    int wave  = threadIdx.x >> 5;
    int lane  = threadIdx.x & 31;
    int mtile = blockIdx.x * 8 + wave;
    if (mtile >= NB / 16) return;
    int hi = lane >> 4, ln = lane & 15;

    const _Float16* arow = h + (size_t)(mtile * 16 + ln) * 64;
    v16h a0 = load_a_frag(arow, 0, hi);
    v16h a1 = load_a_frag(arow, 1, hi);

    float sm[8] = {0.f, 0.f, 0.f, 0.f, 0.f, 0.f, 0.f, 0.f};
    float ss[8] = {0.f, 0.f, 0.f, 0.f, 0.f, 0.f, 0.f, 0.f};
    #pragma unroll
    for (int nt = 0; nt < 16; ++nt) {
        v8f acc = {};
        v16h b0 = *(const v16h*)(bf + nt * 1024 + lane * 16);
        v16h b1 = *(const v16h*)(bf + nt * 1024 + 512 + lane * 16);
        acc = __builtin_amdgcn_wmma_f32_16x16x32_f16(false, a0, false, b0, (short)0, acc, false, false);
        acc = __builtin_amdgcn_wmma_f32_16x16x32_f16(false, a1, false, b1, (short)0, acc, false, false);
        int col = nt * 16 + ln;
        float pb = preb[col], mw = meanw[col], sw = stdw[col];
        #pragma unroll
        for (int r = 0; r < 8; ++r) {
            float p = fmaxf(acc[r] + pb, 0.f);             // relu
            sm[r] += p * mw;
            ss[r] += p * sw;
        }
    }
    // reduce the 16 lanes holding different columns of the same row
    #pragma unroll
    for (int m = 8; m >= 1; m >>= 1) {
        #pragma unroll
        for (int r = 0; r < 8; ++r) {
            sm[r] += __shfl_xor(sm[r], m, 32);
            ss[r] += __shfl_xor(ss[r], m, 32);
        }
    }
    if (ln == 0) {
        float mb = meanb[0], sb = stdb[0];
        #pragma unroll
        for (int r = 0; r < 8; ++r) {
            int row = mtile * 16 + hi * 8 + r;
            float mean = sm[r] + mb;
            float lv   = ss[r] + sb;
            float smp  = noise[RFLD + row] * __expf(0.5f * lv) + mean;
            outp[row]            = smp;    // sample
            outp[NB + row]       = mean;   // mean
            outp[2 * NB + row]   = lv;     // logvar
        }
    }
}

extern "C" void kernel_launch(void* const* d_in, const int* in_sizes, int n_in,
                              void* d_out, int out_size, void* d_ws, size_t ws_size,
                              hipStream_t stream) {
    const float* mgc      = (const float*)d_in[0];
    const float* noise    = (const float*)d_in[1];
    const float* cond_w   = (const float*)d_in[2];
    const float* cond_b   = (const float*)d_in[3];
    const float* cw0_in   = (const float*)d_in[4];
    const float* cw0_gate = (const float*)d_in[5];
    const float* cw0_res  = (const float*)d_in[6];
    const float* cw_in    = (const float*)d_in[7];
    const float* cw_gate  = (const float*)d_in[8];
    const float* cw_res   = (const float*)d_in[9];
    const float* cb_in    = (const float*)d_in[10];
    const float* cb_gate  = (const float*)d_in[11];
    const float* cb_res   = (const float*)d_in[12];
    const float* lw_in    = (const float*)d_in[13];
    const float* lb_in    = (const float*)d_in[14];
    const float* lw_gate  = (const float*)d_in[15];
    const float* lb_gate  = (const float*)d_in[16];
    const float* pre_w    = (const float*)d_in[17];
    const float* pre_b    = (const float*)d_in[18];
    const float* mean_w   = (const float*)d_in[19];
    const float* mean_b   = (const float*)d_in[20];
    const float* std_w    = (const float*)d_in[21];
    const float* std_b    = (const float*)d_in[22];

    char* ws = (char*)d_ws;
    float*     cond  = (float*)ws;                              // 480 KB
    float*     ic    = (float*)(ws + (512 * 1024));             // 10*2000*64 f32
    float*     gc    = ic + NLAY * NB * FSZ;
    _Float16*  wblob = (_Float16*)(ws + (12u * 1024u * 1024u)); // 9*24576 f16
    _Float16*  hblob = wblob + 9 * 24576;                       // 16384 f16
    _Float16*  actA  = (_Float16*)(ws + (16u * 1024u * 1024u)); // 2000*512*64 f16
    _Float16*  actB  = actA + (size_t)NB * 512 * FSZ;           // 2000*256*64 f16

    k_cond<<<(NB * MGCS + 255) / 256, 256, 0, stream>>>(mgc, cond_w, cond_b, cond);
    k_condproj<<<(NLAY * NB * FSZ + 255) / 256, 256, 0, stream>>>(
        cond, lw_in, lb_in, lw_gate, lb_gate, ic, gc);
    k_prepw<<<(9 * 3 * 8192 + 255) / 256, 256, 0, stream>>>(cw_in, cw_gate, cw_res, wblob);
    k_preph<<<(16384 + 255) / 256, 256, 0, stream>>>(pre_w, hblob);
    k_layer0<<<(NB * 512 + 255) / 256, 256, 0, stream>>>(
        noise, cw0_in, cw0_gate, cw0_res, cb_in, cb_gate, cb_res, ic, gc, actA);

    for (int l = 1; l < NLAY; ++l) {
        int Lout   = 512 >> l;
        int mtiles = (NB * Lout) / 16;
        int pairs  = (mtiles + 1) / 2;
        const _Float16* in  = (l & 1) ? actA : actB;
        _Float16*       out = (l & 1) ? actB : actA;
        k_layer<<<(pairs + 7) / 8, 256, 0, stream>>>(
            in, out,
            wblob + (size_t)(l - 1) * 24576,
            cb_in + l * FSZ, cb_gate + l * FSZ, cb_res + l * FSZ,
            ic + (size_t)l * NB * FSZ, gc + (size_t)l * NB * FSZ,
            9 - l, mtiles);
    }

    // layer 9 output (2000 x 64, f16) lives in actB
    k_head<<<16, 256, 0, stream>>>(actB, hblob, pre_b, mean_w, mean_b,
                                   std_w, std_b, noise, (float*)d_out);
}
